// HFLongFormerSelfAttentionBlock_17222818857305
// MI455X (gfx1250) — compile-verified, matrix operand order checked by reading, not run on hardware
//
#include <hip/hip_runtime.h>
#include <hip/hip_bf16.h>
#include <stdint.h>

// ---------------------------------------------------------------------------
// Problem constants (HFLongFormerSelfAttentionBlock, B=2, S0=2048)
// ---------------------------------------------------------------------------
constexpr int BB  = 2;
constexpr int S0  = 2048;
constexpr int SS  = 2304;         // S0 padded to multiple of WIN=256
constexpr int HH  = 1024;
constexpr int NHD = 16;
constexpr int HD  = 64;
constexpr int WW  = 128;          // half window
constexpr int SP  = SS + 2 * WW;  // padded key/value length for vT = 2560
constexpr int HF  = 4096;         // FFN hidden
constexpr int NC  = 16;           // chunks that matter (rows < S0)
constexpr float NEGF = -3.402823466e38f;

// ---------------------------------------------------------------------------
// WMMA types / helpers (gfx1250, wave32)
// ---------------------------------------------------------------------------
typedef __attribute__((ext_vector_type(16))) __bf16 v16bf;
typedef __attribute__((ext_vector_type(8)))  float  v8f;

union FragB16 { v16bf v; uint4 u[2]; };

__device__ inline float bf2f(__bf16 x) {
  unsigned short u = __builtin_bit_cast(unsigned short, x);
  unsigned v = (unsigned)u << 16;
  return __builtin_bit_cast(float, v);
}
__device__ inline __bf16 f2bf(float f) {
  unsigned u = __builtin_bit_cast(unsigned, f);
  unsigned r = (u + 0x7FFFu + ((u >> 16) & 1u)) >> 16;
  return __builtin_bit_cast(__bf16, (unsigned short)r);
}
__device__ inline v8f wmma_bf16(const FragB16& a, const FragB16& b, v8f c) {
  return __builtin_amdgcn_wmma_f32_16x16x32_bf16(false, a.v, false, b.v,
                                                 (short)0, c, false, false);
}

// ---------------------------------------------------------------------------
// Small prep kernels
// ---------------------------------------------------------------------------
__global__ __launch_bounds__(256) void k_maskprep(const int* __restrict__ mask,
                                                  float* __restrict__ fmask,
                                                  float* __restrict__ zrow) {
  int i = blockIdx.x * 256 + threadIdx.x;            // over BB*SS
  int b = i / SS, s = i - b * SS;
  int mv = (s < S0) ? mask[b * S0 + s] : 1;
  float lm = (float)mv * -10000.0f;
  if (s == SS - 1) lm = 10000.0f;
  fmask[i] = (lm != 0.0f) ? NEGF : 0.0f;
  zrow[i]  = (lm < 0.0f) ? 0.0f : 1.0f;
}

__global__ __launch_bounds__(256) void k_xpbf(const float* __restrict__ x,
                                              __bf16* __restrict__ xp) {
  int i = blockIdx.x * 256 + threadIdx.x;            // over BB*SS*HH
  int j = i % HH;
  int s = (i / HH) % SS;
  int b = i / (HH * SS);
  float v = (s < S0) ? x[((size_t)(b * S0 + s)) * HH + j] : 0.0f;
  xp[i] = f2bf(v);
}

// dst[n*K + k] = bf16(src[k*N + n])   (transpose + convert)
__global__ __launch_bounds__(256) void k_tconv(const float* __restrict__ src,
                                               __bf16* __restrict__ dst,
                                               int K, int N) {
  int i = blockIdx.x * 256 + threadIdx.x;            // over K*N
  int k = i / N, n = i - k * N;
  dst[(size_t)n * K + k] = f2bf(src[i]);
}

// vT[b,h,d, sp] = v[b, sp-W, h, d] (zero padded)
__global__ __launch_bounds__(256) void k_vtrans(const __bf16* __restrict__ v,
                                                __bf16* __restrict__ vT) {
  int i = blockIdx.x * 256 + threadIdx.x;            // over BB*NHD*HD*SP
  int sp = i % SP;
  int d  = (i / SP) % HD;
  int h  = (i / (SP * HD)) % NHD;
  int b  = i / (SP * HD * NHD);
  __bf16 val = __builtin_bit_cast(__bf16, (unsigned short)0);
  int s = sp - WW;
  if (s >= 0 && s < SS)
    val = v[(((size_t)(b * SS + s)) * NHD + h) * HD + d];
  vT[i] = val;
}

// ---------------------------------------------------------------------------
// Generic bf16 GEMM: out = epilogue(A[MxK] * Bt[NxK]^T + bias)
// block = 256 (8 waves); block tile 256(M) x 64(N); wave tile 32 x 64
// (two A fragments share four B fragments -> 8 WMMAs per 12 b128 loads)
// MODE 1: bf16 out = (acc+bias)*scale
// MODE 2: bf16 out = gelu_exact(acc+bias)
// MODE 3: f32  out = res + acc + bias
// ---------------------------------------------------------------------------
template <int MODE>
__global__ __launch_bounds__(256) void k_gemm(const __bf16* __restrict__ A,
                                              const __bf16* __restrict__ Bt,
                                              const float* __restrict__ bias,
                                              void* __restrict__ out,
                                              const float* __restrict__ res,
                                              int M, int N, int K, float scale) {
  const int tid  = threadIdx.x;
  const int lane = tid & 31, wid = tid >> 5;
  const int l = lane & 15, hi = lane >> 4;
  const int mrow = blockIdx.x * 256 + wid * 32;
  const int n0b  = blockIdx.y * 64;

  const v8f Z8 = {0.f,0.f,0.f,0.f,0.f,0.f,0.f,0.f};
  v8f acc[2][4] = {{Z8, Z8, Z8, Z8}, {Z8, Z8, Z8, Z8}};

  const __bf16* arow0 = A + (size_t)(mrow + l) * K;
  const __bf16* arow1 = A + (size_t)(mrow + 16 + l) * K;
  for (int k0 = 0; k0 < K; k0 += 32) {
    FragB16 a0, a1;
    a0.u[0] = *(const uint4*)(arow0 + k0 + hi * 8);
    a0.u[1] = *(const uint4*)(arow0 + k0 + 16 + hi * 8);
    a1.u[0] = *(const uint4*)(arow1 + k0 + hi * 8);
    a1.u[1] = *(const uint4*)(arow1 + k0 + 16 + hi * 8);
    // speculative prefetch of the next K-slab (gfx1250 global_prefetch path)
    __builtin_prefetch(arow0 + k0 + 32, 0, 1);
    __builtin_prefetch(arow1 + k0 + 32, 0, 1);
#pragma unroll
    for (int t = 0; t < 4; ++t) {
      const __bf16* brow = Bt + (size_t)(n0b + t * 16 + l) * K + k0;
      FragB16 bb;
      bb.u[0] = *(const uint4*)(brow + hi * 8);
      bb.u[1] = *(const uint4*)(brow + 16 + hi * 8);
      acc[0][t] = wmma_bf16(a0, bb, acc[0][t]);
      acc[1][t] = wmma_bf16(a1, bb, acc[1][t]);
    }
  }
#pragma unroll
  for (int mt = 0; mt < 2; ++mt) {
#pragma unroll
    for (int t = 0; t < 4; ++t) {
      const int n = n0b + t * 16 + l;
      const float bn = bias[n];
#pragma unroll
      for (int r = 0; r < 8; ++r) {
        const int m = mrow + mt * 16 + r + 8 * hi;
        float val = acc[mt][t][r] + bn;
        if (MODE == 1) {
          ((__bf16*)out)[(size_t)m * N + n] = f2bf(val * scale);
        } else if (MODE == 2) {
          float g = 0.5f * val * (1.0f + erff(val * 0.70710678118654752f));
          ((__bf16*)out)[(size_t)m * N + n] = f2bf(g);
        } else {
          ((float*)out)[(size_t)m * N + n] = res[(size_t)m * N + n] + val;
        }
      }
    }
  }
}

// ---------------------------------------------------------------------------
// Fused banded attention: one block per (b, h, chunk c<16), 8 waves.
// LDS: sc f32[128][384] | pc bf16[128][392] | gp f32[128]   (~291 KB)
// Each wave owns rows [wid*16, wid*16+16) end-to-end -> no barriers needed.
// Out-of-window key rows are loaded clamped-to-0 (garbage): those score
// columns are provably never read (softmax applies the j-in-range test before
// touching sc), so no branch / zero-fill is needed on the load path.
// ---------------------------------------------------------------------------
constexpr int SC_BYTES = 128 * 384 * 4;
constexpr int PC_BYTES = 128 * 392 * 2;
constexpr int SMEM_BYTES = SC_BYTES + PC_BYTES + 128 * 4;

__global__ __launch_bounds__(256) void k_attn(const __bf16* __restrict__ q,
                                              const __bf16* __restrict__ k,
                                              const __bf16* __restrict__ v,
                                              const __bf16* __restrict__ vT,
                                              const float* __restrict__ fmask,
                                              const float* __restrict__ zrow,
                                              float* __restrict__ attn) {
  extern __shared__ char smem[];
  float*  sc = (float*)smem;
  __bf16* pc = (__bf16*)(smem + SC_BYTES);
  float*  gp = (float*)(smem + SC_BYTES + PC_BYTES);

  const int tid  = threadIdx.x;
  const int lane = tid & 31, wid = tid >> 5;
  const int l = lane & 15, hi = lane >> 4;
  int idx = blockIdx.x;
  const int c = idx & 15; idx >>= 4;
  const int h = idx & 15; const int b = idx >> 4;
  const int cW = c * WW;
  const int m0 = wid * 16;
  const v8f Z8 = {0.f,0.f,0.f,0.f,0.f,0.f,0.f,0.f};

  // ---------------- Phase 1: scores sc = Qc (128x64) * Kc^T (64x384) --------
  {
    const __bf16* qrow = q + (((size_t)(b * SS + cW + m0 + l)) * NHD + h) * HD;
    FragB16 a0, a1;
    a0.u[0] = *(const uint4*)(qrow + hi * 8);
    a0.u[1] = *(const uint4*)(qrow + 16 + hi * 8);
    a1.u[0] = *(const uint4*)(qrow + 32 + hi * 8);
    a1.u[1] = *(const uint4*)(qrow + 48 + hi * 8);
    for (int nt = 0; nt < 24; ++nt) {
      const int n0 = nt * 16;
      int sj = cW + n0 + l - WW;                // key row; sj < SS always here
      sj = (sj < 0) ? 0 : sj;                   // clamp: dead columns, no guard
      const __bf16* krow = k + (((size_t)(b * SS + sj)) * NHD + h) * HD;
      FragB16 b0, b1;
      b0.u[0] = *(const uint4*)(krow + hi * 8);
      b0.u[1] = *(const uint4*)(krow + 16 + hi * 8);
      b1.u[0] = *(const uint4*)(krow + 32 + hi * 8);
      b1.u[1] = *(const uint4*)(krow + 48 + hi * 8);
      v8f acc = Z8;
      acc = wmma_bf16(a0, b0, acc);
      acc = wmma_bf16(a1, b1, acc);
#pragma unroll
      for (int r = 0; r < 8; ++r)
        sc[(m0 + r + 8 * hi) * 384 + n0 + l] = acc[r];
    }
  }

  // ---------------- Phase 2: per-row softmax over [gcol, band(257)] ---------
  {
    const __bf16* klast = k + (((size_t)(b * SS + (SS - 1))) * NHD + h) * HD;
    const float NEGINF = -__builtin_inff();
    for (int p = 0; p < 16; ++p) {
      const int P  = m0 + p;
      const int sg = cW + P;
      const __bf16* qrow = q + (((size_t)(b * SS + sg)) * NHD + h) * HD;
      float part = bf2f(qrow[lane]) * bf2f(klast[lane]) +
                   bf2f(qrow[lane + 32]) * bf2f(klast[lane + 32]);
      for (int off = 16; off; off >>= 1) part += __shfl_xor(part, off, 32);
      const float gdot = part;

      float vals[9];
      float mx = gdot;
#pragma unroll
      for (int i = 0; i < 9; ++i) {
        const int t = lane + 32 * i;
        float vv = NEGINF;
        if (t < 257) {
          const int j = sg + t - WW;
          if (j >= 0 && j < SS) vv = sc[P * 384 + (P + t)] + fmask[b * SS + j];
        }
        vals[i] = vv;
        mx = fmaxf(mx, vv);
      }
      for (int off = 16; off; off >>= 1) mx = fmaxf(mx, __shfl_xor(mx, off, 32));
      float ss = 0.f;
#pragma unroll
      for (int i = 0; i < 9; ++i) ss += __expf(vals[i] - mx);
      for (int off = 16; off; off >>= 1) ss += __shfl_xor(ss, off, 32);
      ss += __expf(gdot - mx);                  // global column term (all lanes)
      const float z   = zrow[b * SS + sg];
      const float inv = z / ss;
      if (lane == 0) gp[P] = __expf(gdot - mx) * inv;

      for (int col = lane; col < 384; col += 32) {
        const int t = col - P;
        float pv = 0.f;
        if (t >= 0 && t <= 256) {
          const int j = sg + t - WW;
          if (j >= 0 && j < SS)
            pv = __expf(sc[P * 384 + col] + fmask[b * SS + j] - mx) * inv;
        }
        pc[P * 392 + col] = f2bf(pv);
      }
    }
  }

  // ---------------- Phase 3: out_l = P (128x384) * Vc (384x64) + g*v_last ---
  {
    const __bf16* vl = v + (((size_t)(b * SS + (SS - 1))) * NHD + h) * HD;
    v8f acc[4] = {Z8, Z8, Z8, Z8};
    const __bf16* prow = pc + (m0 + l) * 392;
    for (int k0 = 0; k0 < 384; k0 += 32) {
      FragB16 a;
      a.u[0] = *(const uint4*)(prow + k0 + hi * 8);
      a.u[1] = *(const uint4*)(prow + k0 + 16 + hi * 8);
#pragma unroll
      for (int t = 0; t < 4; ++t) {
        const __bf16* vrow = vT + ((size_t)((b * NHD + h) * HD) + t * 16 + l) * SP
                                 + cW + k0;
        FragB16 bb;
        bb.u[0] = *(const uint4*)(vrow + hi * 8);
        bb.u[1] = *(const uint4*)(vrow + 16 + hi * 8);
        acc[t] = wmma_bf16(a, bb, acc[t]);
      }
    }
#pragma unroll
    for (int t = 0; t < 4; ++t) {
      const float vv = bf2f(vl[t * 16 + l]);
#pragma unroll
      for (int r = 0; r < 8; ++r) {
        const int P = m0 + r + 8 * hi;
        const int s = cW + P;                   // < S0 guaranteed (c < 16)
        attn[((size_t)(b * S0 + s)) * HH + h * HD + t * 16 + l] =
            acc[t][r] + gp[P] * vv;
      }
    }
  }
}

// ---------------------------------------------------------------------------
// LayerNorm: y = LN(x + attn) * g + b   (one row per block)
// ---------------------------------------------------------------------------
__global__ __launch_bounds__(256) void k_ln(const float* __restrict__ x,
                                            const float* __restrict__ attn,
                                            const float* __restrict__ g,
                                            const float* __restrict__ be,
                                            float* __restrict__ yln,
                                            __bf16* __restrict__ ybf) {
  const int row = blockIdx.x;
  const int tid = threadIdx.x;
  __shared__ float red[8];
  float vals[4];
  float s1 = 0.f;
#pragma unroll
  for (int i = 0; i < 4; ++i) {
    const int j = tid + i * 256;
    const float vv = x[(size_t)row * HH + j] + attn[(size_t)row * HH + j];
    vals[i] = vv; s1 += vv;
  }
  for (int off = 16; off; off >>= 1) s1 += __shfl_xor(s1, off, 32);
  if ((tid & 31) == 0) red[tid >> 5] = s1;
  __syncthreads();
  float tot = 0.f;
  for (int w = 0; w < 8; ++w) tot += red[w];
  const float mu = tot * (1.0f / HH);
  __syncthreads();
  float s2 = 0.f;
#pragma unroll
  for (int i = 0; i < 4; ++i) { float d = vals[i] - mu; s2 += d * d; }
  for (int off = 16; off; off >>= 1) s2 += __shfl_xor(s2, off, 32);
  if ((tid & 31) == 0) red[tid >> 5] = s2;
  __syncthreads();
  float vtot = 0.f;
  for (int w = 0; w < 8; ++w) vtot += red[w];
  const float rs = rsqrtf(vtot * (1.0f / HH) + 1e-5f);
#pragma unroll
  for (int i = 0; i < 4; ++i) {
    const int j = tid + i * 256;
    const float o = (vals[i] - mu) * rs * g[j] + be[j];
    yln[(size_t)row * HH + j] = o;
    ybf[(size_t)row * HH + j] = f2bf(o);
  }
}

// ---------------------------------------------------------------------------
// Host-side orchestration
// ---------------------------------------------------------------------------
extern "C" void kernel_launch(void* const* d_in, const int* in_sizes, int n_in,
                              void* d_out, int out_size, void* d_ws, size_t ws_size,
                              hipStream_t stream) {
  (void)in_sizes; (void)n_in; (void)out_size; (void)ws_size;
  const float* x    = (const float*)d_in[0];
  const int*   mask = (const int*)d_in[1];
  const float* wq = (const float*)d_in[2];  const float* bq = (const float*)d_in[3];
  const float* wk = (const float*)d_in[4];  const float* bk = (const float*)d_in[5];
  const float* wv = (const float*)d_in[6];  const float* bv = (const float*)d_in[7];
  // d_in[8..13] = wqg,bqg,wkg,bkg,wvg,bvg : dead code (last row is sliced off)
  const float* ln_g = (const float*)d_in[14];
  const float* ln_b = (const float*)d_in[15];
  const float* w1 = (const float*)d_in[16]; const float* b1 = (const float*)d_in[17];
  const float* w2 = (const float*)d_in[18]; const float* b2 = (const float*)d_in[19];
  float* out = (float*)d_out;

  char* ws = (char*)d_ws;
  size_t off = 0;
  auto alloc = [&](size_t bytes) -> char* {
    char* p = ws + off;
    off = (off + bytes + 255) & ~(size_t)255;
    return p;
  };
  __bf16* xp_bf = (__bf16*)alloc((size_t)BB * SS * HH * 2);
  __bf16* wq_t  = (__bf16*)alloc((size_t)HH * HH * 2);
  __bf16* wk_t  = (__bf16*)alloc((size_t)HH * HH * 2);
  __bf16* wv_t  = (__bf16*)alloc((size_t)HH * HH * 2);
  __bf16* w1_t  = (__bf16*)alloc((size_t)HH * HF * 2);
  __bf16* w2_t  = (__bf16*)alloc((size_t)HF * HH * 2);
  __bf16* q_bf  = (__bf16*)alloc((size_t)BB * SS * HH * 2);
  __bf16* k_bf  = (__bf16*)alloc((size_t)BB * SS * HH * 2);
  __bf16* v_bf  = (__bf16*)alloc((size_t)BB * SS * HH * 2);
  __bf16* vT    = (__bf16*)alloc((size_t)BB * NHD * HD * SP * 2);
  float*  fmask = (float*)alloc((size_t)BB * SS * 4);
  float*  zrow  = (float*)alloc((size_t)BB * SS * 4);
  float*  attn  = (float*)alloc((size_t)BB * S0 * HH * 4);
  float*  yln   = (float*)alloc((size_t)BB * S0 * HH * 4);
  __bf16* ybf   = (__bf16*)alloc((size_t)BB * S0 * HH * 2);
  __bf16* h_bf  = (__bf16*)alloc((size_t)BB * S0 * HF * 2);

  // prep
  k_maskprep<<<(BB * SS) / 256, 256, 0, stream>>>(mask, fmask, zrow);
  k_xpbf<<<(BB * SS * HH) / 256, 256, 0, stream>>>(x, xp_bf);
  k_tconv<<<(HH * HH) / 256, 256, 0, stream>>>(wq, wq_t, HH, HH);
  k_tconv<<<(HH * HH) / 256, 256, 0, stream>>>(wk, wk_t, HH, HH);
  k_tconv<<<(HH * HH) / 256, 256, 0, stream>>>(wv, wv_t, HH, HH);
  k_tconv<<<(HH * HF) / 256, 256, 0, stream>>>(w1, w1_t, HH, HF);
  k_tconv<<<(HF * HH) / 256, 256, 0, stream>>>(w2, w2_t, HF, HH);

  // QKV projections (bf16 WMMA, f32 accumulate)
  const int Mx = BB * SS;  // 4608
  dim3 gq(Mx / 256, HH / 64);
  k_gemm<1><<<gq, 256, 0, stream>>>(xp_bf, wq_t, bq, (void*)q_bf, nullptr,
                                    Mx, HH, HH, 0.125f);
  k_gemm<1><<<gq, 256, 0, stream>>>(xp_bf, wk_t, bk, (void*)k_bf, nullptr,
                                    Mx, HH, HH, 1.0f);
  k_gemm<1><<<gq, 256, 0, stream>>>(xp_bf, wv_t, bv, (void*)v_bf, nullptr,
                                    Mx, HH, HH, 1.0f);
  k_vtrans<<<(BB * NHD * HD * SP) / 256, 256, 0, stream>>>(v_bf, vT);

  // fused banded attention (needs 291 KB dynamic LDS; CDNA5 WGP has 320 KB)
  (void)hipFuncSetAttribute((const void*)k_attn,
                            hipFuncAttributeMaxDynamicSharedMemorySize,
                            SMEM_BYTES);
  k_attn<<<BB * NHD * NC, 256, SMEM_BYTES, stream>>>(q_bf, k_bf, v_bf, vT,
                                                     fmask, zrow, attn);

  // y = LN(x + attn)
  k_ln<<<BB * S0, 256, 0, stream>>>(x, attn, ln_g, ln_b, yln, ybf);

  // FFN: h = gelu(y @ w1 + b1); out = y + h @ w2 + b2
  const int Mf = BB * S0;  // 4096
  dim3 g1(Mf / 256, HF / 64);
  k_gemm<2><<<g1, 256, 0, stream>>>(ybf, w1_t, b1, (void*)h_bf, nullptr,
                                    Mf, HF, HH, 1.0f);
  dim3 g2(Mf / 256, HH / 64);
  k_gemm<3><<<g2, 256, 0, stream>>>(h_bf, w2_t, b2, (void*)out, yln,
                                    Mf, HH, HF, 1.0f);
}
